// rnn_model_74612171866343
// MI455X (gfx1250) — matrix operand order repaired and checked by reading, not compiled, not run on hardware
//
#include <hip/hip_runtime.h>

typedef __attribute__((ext_vector_type(16))) _Float16 v16h;
typedef __attribute__((ext_vector_type(8)))  _Float16 v8h;
typedef __attribute__((ext_vector_type(4)))  _Float16 v4h;
typedef __attribute__((ext_vector_type(8)))  float    v8f;

#define N_IN  64
#define N_H   128
#define N_OUT 32

// Workspace layout (bytes):
//   [0,      16384)  B1 blob f16: GEMM1 B fragments (natural K order).
//   [16384,  24576)  B2 blob f16: GEMM2 B fragments (PERMUTED K order:
//                    storage position P -> neuron n(P) = (P&7)*16 + (P>>3)).
//   [24576,  25088)  driveP f32[128], permuted: driveP[(n&15)*8 + (n>>4)] = drive[n]
#define WS_B2_OFF   16384
#define WS_DRV_OFF  24576

// ---------------- Prep kernel: weight reformat + hidden drive -----------------
__global__ __launch_bounds__(256) void rnn_prep_74612171866343(
    const float* __restrict__ W_ih, const float* __restrict__ W_hh,
    const float* __restrict__ b_ih, const float* __restrict__ b_hh,
    const float* __restrict__ W_out, const float* __restrict__ h0,
    _Float16* __restrict__ blobB1, _Float16* __restrict__ blobB2,
    float* __restrict__ driveP)
{
    const int tid = threadIdx.x;

    // drive[n] = b_ih[n] + b_hh[n] + sum_k h0[k] * W_hh[n][k], stored permuted
    if (tid < N_H) {
        float s = b_ih[tid] + b_hh[tid];
        const float* wr = W_hh + tid * N_H;
        #pragma unroll 8
        for (int k = 0; k < N_H; ++k) s += wr[k] * h0[k];
        driveP[(tid & 15) * 8 + (tid >> 4)] = s;
    }

    // B1 fragments (natural K): b[j] = W_ih^T[kt*32 + (j>>3)*16 + hi*8 + (j&7)][nt*16 + (lane&15)]
    for (int t = tid; t < 16 * 32 * 16; t += 256) {
        int j    = t & 15;
        int lane = (t >> 4) & 31;
        int frag = t >> 9;            // 0..15
        int kt   = frag & 1;
        int nt   = frag >> 1;
        int hi   = lane >> 4;
        int n    = nt * 16 + (lane & 15);
        int k    = kt * 32 + (j >> 3) * 16 + hi * 8 + (j & 7);
        blobB1[t] = (_Float16)W_ih[n * N_IN + k];   // W_ih is [N_H][N_IN] row-major
    }

    // B2 fragments (permuted K): storage position P = kt*32 + (j>>3)*16 + hi*8 + (j&7) maps to
    // neuron n(P) = (P&7)*16 + (P>>3); b[j] = W_out[n(P)][nt*16 + (lane&15)]
    for (int t = tid; t < 8 * 32 * 16; t += 256) {
        int j    = t & 15;
        int lane = (t >> 4) & 31;
        int frag = t >> 9;            // 0..7
        int kt   = frag & 3;
        int nt   = frag >> 2;
        int hi   = lane >> 4;
        int nout = nt * 16 + (lane & 15);
        int P    = kt * 32 + (j >> 3) * 16 + hi * 8 + (j & 7);
        int kn   = (P & 7) * 16 + (P >> 3);
        blobB2[t] = (_Float16)W_out[kn * N_OUT + nout]; // W_out is [N_H][N_OUT]
    }
}

// ---------------- Main kernel: fused GEMM1(relu) -> GEMM2 ---------------------
// Each wave owns 32 rows of T (two 16-row M-tiles); every B fragment is loaded
// once and fed to both M-tiles' WMMAs.
__global__ __launch_bounds__(256) void rnn_main_74612171866343(
    const float* __restrict__ x,
    const _Float16* __restrict__ blobB1,
    const _Float16* __restrict__ blobB2,
    const float* __restrict__ driveP,
    const float* __restrict__ b_out,
    float* __restrict__ out)
{
    // Per-wave private 32x128 f16 h-tile, K stored in permuted order sigma(n).
    __shared__ _Float16 hbuf[8][32][N_H];   // 64 KB / block

    const int wave = threadIdx.x >> 5;
    const int lane = threadIdx.x & 31;
    const int hi   = lane >> 4;        // half-wave select
    const int m    = lane & 15;        // A-row / C-col index
    const size_t rowBase = (size_t)blockIdx.x * 256 + (size_t)wave * 32;

    // ---- A fragments for GEMM1 (f16 A 16x32 layout), both M-tiles:
    // lane holds K in [hi*8, hi*8+8) -> a[0..7], and [16+hi*8, ...) -> a[8..15]
    v16h a1[2][2];
    #pragma unroll
    for (int mt = 0; mt < 2; ++mt) {
        const float* xr = x + (rowBase + (size_t)(mt * 16 + m)) * N_IN;
        #pragma unroll
        for (int kt = 0; kt < 2; ++kt) {
            const float4* p = (const float4*)(xr + kt * 32 + hi * 8);
            float4 r0 = p[0], r1 = p[1];
            const float4* q = (const float4*)(xr + kt * 32 + 16 + hi * 8);
            float4 r2 = q[0], r3 = q[1];
            v16h a;
            a[0]  = (_Float16)r0.x; a[1]  = (_Float16)r0.y; a[2]  = (_Float16)r0.z; a[3]  = (_Float16)r0.w;
            a[4]  = (_Float16)r1.x; a[5]  = (_Float16)r1.y; a[6]  = (_Float16)r1.z; a[7]  = (_Float16)r1.w;
            a[8]  = (_Float16)r2.x; a[9]  = (_Float16)r2.y; a[10] = (_Float16)r2.z; a[11] = (_Float16)r2.w;
            a[12] = (_Float16)r3.x; a[13] = (_Float16)r3.y; a[14] = (_Float16)r3.z; a[15] = (_Float16)r3.w;
            a1[mt][kt] = a;
        }
    }

    // ---- GEMM1: h = relu(X @ W_ih^T + drive), two groups of 4 N-tiles
    const float* dp = driveP + m * 8;   // contiguous 32B: drive for nt = 0..7
    #pragma unroll
    for (int g = 0; g < 2; ++g) {
        v8f acc[2][4];
        #pragma unroll
        for (int i = 0; i < 4; ++i) {
            float dn = dp[g * 4 + i];
            acc[0][i] = (v8f){dn, dn, dn, dn, dn, dn, dn, dn};
            acc[1][i] = (v8f){dn, dn, dn, dn, dn, dn, dn, dn};
        }
        #pragma unroll
        for (int i = 0; i < 4; ++i) {
            const int nt = g * 4 + i;
            #pragma unroll
            for (int kt = 0; kt < 2; ++kt) {
                v16h b = *(const v16h*)(blobB1 + (size_t)(((nt << 1) | kt) * 32 + lane) * 16);
                acc[0][i] = __builtin_amdgcn_wmma_f32_16x16x32_f16(
                    false, a1[0][kt], false, b, (short)0, acc[0][i], false, false);
                acc[1][i] = __builtin_amdgcn_wmma_f32_16x16x32_f16(
                    false, a1[1][kt], false, b, (short)0, acc[1][i], false, false);
            }
        }
        // Convert (v_cvt_pk_f16_f32) then ReLU as packed f16 max
        // (max(cvt(x),0) == cvt(max(x,0)) since RNE cvt is monotonic, cvt(0)=0).
        // Group's 4 nt-values are contiguous at permuted-K positions
        // [m*8 + g*4, +4)  ->  one ds_store_b64 per (mt, r).
        #pragma unroll
        for (int mt = 0; mt < 2; ++mt) {
            #pragma unroll
            for (int r = 0; r < 8; ++r) {
                v4h pk;
                #pragma unroll
                for (int i = 0; i < 4; ++i)
                    pk[i] = (_Float16)acc[mt][i][r];
                v4h z = {};
                pk = __builtin_elementwise_max(pk, z);   // v_pk_max_num_f16 x2
                *(v4h*)(&hbuf[wave][mt * 16 + r + hi * 8][m * 8 + g * 4]) = pk;
            }
        }
    }

    // ---- Rebuild A fragments for GEMM2 from LDS (two ds_load_b128 per K-tile)
    v16h a2[2][4];
    #pragma unroll
    for (int mt = 0; mt < 2; ++mt) {
        const _Float16* hr = &hbuf[wave][mt * 16 + m][0];
        #pragma unroll
        for (int kt = 0; kt < 4; ++kt) {
            v8h lo = *(const v8h*)(hr + kt * 32 + hi * 8);
            v8h hh = *(const v8h*)(hr + kt * 32 + 16 + hi * 8);
            v16h a;
            #pragma unroll
            for (int j = 0; j < 8; ++j) { a[j] = lo[j]; a[j + 8] = hh[j]; }
            a2[mt][kt] = a;
        }
    }

    // ---- GEMM2: out = h @ W_out + b_out (permuted K), B shared across M-tiles
    #pragma unroll
    for (int nt = 0; nt < 2; ++nt) {
        float bo = b_out[nt * 16 + m];
        v8f acc2[2];
        acc2[0] = (v8f){bo, bo, bo, bo, bo, bo, bo, bo};
        acc2[1] = (v8f){bo, bo, bo, bo, bo, bo, bo, bo};
        #pragma unroll
        for (int kt = 0; kt < 4; ++kt) {
            v16h b = *(const v16h*)(blobB2 + (size_t)(((nt << 2) | kt) * 32 + lane) * 16);
            acc2[0] = __builtin_amdgcn_wmma_f32_16x16x32_f16(
                false, a2[0][kt], false, b, (short)0, acc2[0], false, false);
            acc2[1] = __builtin_amdgcn_wmma_f32_16x16x32_f16(
                false, a2[1][kt], false, b, (short)0, acc2[1], false, false);
        }
        #pragma unroll
        for (int mt = 0; mt < 2; ++mt) {
            #pragma unroll
            for (int r = 0; r < 8; ++r) {
                out[(rowBase + (size_t)(mt * 16 + r + hi * 8)) * N_OUT + nt * 16 + m] = acc2[mt][r];
            }
        }
    }
}

extern "C" void kernel_launch(void* const* d_in, const int* in_sizes, int n_in,
                              void* d_out, int out_size, void* d_ws, size_t ws_size,
                              hipStream_t stream) {
    const float* x     = (const float*)d_in[0];
    const float* W_ih  = (const float*)d_in[1];
    const float* W_hh  = (const float*)d_in[2];
    const float* b_ih  = (const float*)d_in[3];
    const float* b_hh  = (const float*)d_in[4];
    const float* W_out = (const float*)d_in[5];
    const float* b_out = (const float*)d_in[6];
    const float* h0    = (const float*)d_in[7];
    float* out = (float*)d_out;

    char* ws = (char*)d_ws;
    _Float16* blobB1 = (_Float16*)ws;
    _Float16* blobB2 = (_Float16*)(ws + WS_B2_OFF);
    float*    driveP = (float*)(ws + WS_DRV_OFF);

    rnn_prep_74612171866343<<<1, 256, 0, stream>>>(
        W_ih, W_hh, b_ih, b_hh, W_out, h0, blobB1, blobB2, driveP);

    const int T = in_sizes[0] / N_IN;        // 1048576 (multiple of 256)
    const int grid = T / 256;                // 4096 blocks, 8 waves x 32 rows
    rnn_main_74612171866343<<<grid, 256, 0, stream>>>(
        x, blobB1, blobB2, driveP, b_out, out);
}
